// SelfAttentionMultiHead_9878424781060
// MI455X (gfx1250) — compile-verified, matrix operand order checked by reading, not run on hardware
//
#include <hip/hip_runtime.h>

typedef __attribute__((ext_vector_type(16))) _Float16 v16h;
typedef __attribute__((ext_vector_type(8)))  _Float16 v8h;
typedef __attribute__((ext_vector_type(8)))  float    v8f;
typedef __attribute__((ext_vector_type(4)))  unsigned int v4u;
typedef __attribute__((ext_vector_type(8)))  int      v8i;
typedef __attribute__((ext_vector_type(4)))  int      v4i;

#define D_MODEL 1024
#define N_HEADS 16
#define D_HEAD  64
#define BATCH   2
#define SEQ     2048
#define M_TOT   (BATCH * SEQ)   // 4096

#if defined(__has_builtin)
#if __has_builtin(__builtin_amdgcn_tensor_load_to_lds)
#define USE_TDM 1
#endif
#endif

// ---------------------------------------------------------------- helpers

__device__ __forceinline__ v8f vzero() {
  v8f z = {0.f, 0.f, 0.f, 0.f, 0.f, 0.f, 0.f, 0.f};
  return z;
}

__device__ __forceinline__ v8f wmma16(v16h a, v16h b, v8f c) {
  return __builtin_amdgcn_wmma_f32_16x16x32_f16(
      false, a, false, b, (short)0, c, false, false);
}

// Same WMMA but hinting that B is identical to the previous instruction's B
// (ISA 7.12: RB reuse legal when current instruction matches previous).
__device__ __forceinline__ v8f wmma16_rb(v16h a, v16h b, v8f c) {
  return __builtin_amdgcn_wmma_f32_16x16x32_f16(
      false, a, false, b, (short)0, c, false, true);
}

// Load a 16-bit A/B fragment from a row-major, K-contiguous matrix.
// Per ISA layout: lane = row, half-wave selects K base +8; lane holds
// K = kb..kb+7 (VGPR 0..3) and K = kb+16..kb+23 (VGPR 4..7).
__device__ __forceinline__ v16h load_frag(const _Float16* __restrict__ base,
                                          int row, int ld, int kBase, int lane) {
  const int hf = lane >> 4;
  const _Float16* p = base + (size_t)row * ld + kBase + hf * 8;
  v16h a;
  *(v8h*)&a       = *(const v8h*)(p);
  *((v8h*)&a + 1) = *(const v8h*)(p + 16);
  return a;
}

__device__ __forceinline__ v16h load_frag_lds(const _Float16* p) {
  v16h a;
  *(v8h*)&a       = *(const v8h*)(p);
  *((v8h*)&a + 1) = *(const v8h*)(p + 16);
  return a;
}

#ifdef USE_TDM
// TDM: DMA a til1-rows x til0-cols f16 tile (row stride stride0 elems) to LDS.
// D# per CDNA5 ISA (group0 + group1, 2D tensor).
__device__ __forceinline__ void tdm_load_2d(unsigned lds_off, const void* gptr,
                                            unsigned tdim0, unsigned tdim1,
                                            unsigned stride0, unsigned til0,
                                            unsigned til1) {
  unsigned long long ga = (unsigned long long)(size_t)gptr;
  v4u g0;
  g0[0] = 1u;                                        // count=1, user mode
  g0[1] = lds_off;                                   // lds_addr (bytes)
  g0[2] = (unsigned)ga;                              // global_addr[31:0]
  g0[3] = (unsigned)((ga >> 32) & 0x01FFFFFFu)       // global_addr[56:32]
          | (2u << 30);                              // type = 2 ("image")
  v8i g1;
  g1[0] = (int)(1u << 16);                           // data_size = 1 -> 2B
  g1[1] = (int)(tdim0 << 16);                        // tensor_dim0[15:0]
  g1[2] = (int)((tdim0 >> 16) | (tdim1 << 16));      // dim0[31:16] dim1[15:0]
  g1[3] = (int)((tdim1 >> 16) | (til0 << 16));       // dim1[31:16] tile_dim0
  g1[4] = (int)(til1 & 0xFFFFu);                     // tile_dim1
  g1[5] = (int)stride0;                              // tensor_dim0_stride
  g1[6] = 0;
  g1[7] = 0;
  v4i z4 = {0, 0, 0, 0};
#if defined(__clang_major__) && (__clang_major__ >= 23)
  v8i z8 = {0, 0, 0, 0, 0, 0, 0, 0};
  __builtin_amdgcn_tensor_load_to_lds(g0, g1, z4, z4, z8, 0);
#else
  __builtin_amdgcn_tensor_load_to_lds(g0, g1, z4, z4, 0);
#endif
}
#else
__device__ __forceinline__ void coop_copy_tile(_Float16* dst,
                                               const _Float16* src,
                                               int rows, int cols, int ld,
                                               int tid, int nthr) {
  for (int i = tid; i < rows * cols; i += nthr) {
    int r = i / cols, c = i % cols;
    dst[i] = src[(size_t)r * ld + c];
  }
}
#endif

// ---------------------------------------------------------------- f32 -> f16

__global__ void f32_to_f16(const float* __restrict__ src,
                           _Float16* __restrict__ dst, int n) {
  int i = blockIdx.x * blockDim.x + threadIdx.x;
  if (i < n) dst[i] = (_Float16)src[i];
}

// ---------------------------------------------------------------- QKV GEMM
// C[m,n] = sum_k Xh[m,k] * W[n,k]   (y = X W^T)
// Block: 256 thr = 8 waves, tile 64(M) x 128(N); wave tile 64x16.
// Both operands DMA'd into LDS by the TDM, double-buffered over K.
// grid = (M/64, N/128, 3 weights)

__global__ __launch_bounds__(256) void qkv_gemm(
    const _Float16* __restrict__ Xh, const _Float16* __restrict__ Wh,
    _Float16* __restrict__ Qh, _Float16* __restrict__ Kh,
    _Float16* __restrict__ Vt) {
  __shared__ _Float16 xtile[2][64 * 32];    // 2 x 4 KB
  __shared__ _Float16 wtile[2][128 * 32];   // 2 x 8 KB
  const int lane = threadIdx.x & 31;
  const int w    = threadIdx.x >> 5;
  const int lm   = lane & 15;
  const int hf   = lane >> 4;
  const int wid  = blockIdx.z;                 // 0=Q 1=K 2=V
  const int mBase = blockIdx.x * 64;
  const int nBlk  = blockIdx.y * 128;
  const int nBase = nBlk + w * 16;
  const _Float16* Wb   = Wh + (size_t)wid * D_MODEL * D_MODEL;
  const _Float16* Xorg = Xh + (size_t)mBase * D_MODEL;
  const _Float16* Worg = Wb + (size_t)nBlk * D_MODEL;

  const int NK = D_MODEL / 32;                 // 32 K-tiles

#ifdef USE_TDM
  const unsigned xb[2] = {(unsigned)(size_t)(&xtile[0][0]),
                          (unsigned)(size_t)(&xtile[1][0])};
  const unsigned wb[2] = {(unsigned)(size_t)(&wtile[0][0]),
                          (unsigned)(size_t)(&wtile[1][0])};
  if (w == 0) {
    tdm_load_2d(xb[0], Xorg, D_MODEL, M_TOT,   D_MODEL, 32, 64);
    tdm_load_2d(wb[0], Worg, D_MODEL, D_MODEL, D_MODEL, 32, 128);
  }
#else
  coop_copy_tile(xtile[0], Xorg, 64, 32, D_MODEL, threadIdx.x, 256);
  coop_copy_tile(wtile[0], Worg, 128, 32, D_MODEL, threadIdx.x, 256);
#endif

  v8f acc[4] = {vzero(), vzero(), vzero(), vzero()};
  for (int kt = 0; kt < NK; ++kt) {
    __syncthreads();   // everyone done reading the buffer being refilled
#ifdef USE_TDM
    if (w == 0) {
      if (kt + 1 < NK) {
        tdm_load_2d(xb[(kt + 1) & 1], Xorg + (kt + 1) * 32,
                    D_MODEL, M_TOT, D_MODEL, 32, 64);
        tdm_load_2d(wb[(kt + 1) & 1], Worg + (kt + 1) * 32,
                    D_MODEL, D_MODEL, D_MODEL, 32, 128);
        __builtin_amdgcn_s_wait_tensorcnt(2);  // current pair complete
      } else {
        __builtin_amdgcn_s_wait_tensorcnt(0);
      }
    }
#else
    if (kt + 1 < NK) {
      coop_copy_tile(xtile[(kt + 1) & 1], Xorg + (kt + 1) * 32,
                     64, 32, D_MODEL, threadIdx.x, 256);
      coop_copy_tile(wtile[(kt + 1) & 1], Worg + (kt + 1) * 32,
                     128, 32, D_MODEL, threadIdx.x, 256);
    }
#endif
    __syncthreads();   // current tiles visible
    const _Float16* xt = xtile[kt & 1];
    const v16h bfrag = load_frag_lds(wtile[kt & 1] + (w * 16 + lm) * 32 + hf * 8);
    acc[0] = wmma16(load_frag_lds(xt + lm * 32 + hf * 8), bfrag, acc[0]);
#pragma unroll
    for (int s = 1; s < 4; ++s) {
      const v16h afrag = load_frag_lds(xt + (s * 16 + lm) * 32 + hf * 8);
      acc[s] = wmma16_rb(afrag, bfrag, acc[s]);  // same B as previous WMMA
    }
  }

  if (wid == 2) {
    // V stored transposed: Vt[(b*H + h)*64 + d][t]; 8 consecutive t per lane
    const int n  = nBase + lm;
    const int hh = n >> 6, d = n & 63;
#pragma unroll
    for (int s = 0; s < 4; ++s) {
      const int m0 = mBase + s * 16 + hf * 8;
      const int bb = m0 >> 11, t0 = m0 & (SEQ - 1);
      v8h pk;
#pragma unroll
      for (int r = 0; r < 8; ++r) pk[r] = (_Float16)acc[s][r];
      *(v8h*)(Vt + ((size_t)(bb * N_HEADS + hh) * D_HEAD + d) * SEQ + t0) = pk;
    }
  } else {
    _Float16* Dst = (wid == 0) ? Qh : Kh;
    const float scl = (wid == 0) ? 0.125f : 1.0f;  // fold 1/sqrt(Dh) into Q
    const int n = nBase + lm;
#pragma unroll
    for (int s = 0; s < 4; ++s)
#pragma unroll
      for (int r = 0; r < 8; ++r) {
        const int m = mBase + s * 16 + hf * 8 + r;
        Dst[(size_t)m * D_MODEL + n] = (_Float16)(acc[s][r] * scl);
      }
  }
}

// ---------------------------------------------------------------- attention
// Block = 8 waves = 128 q rows of one (b,h). K/V key-block tiles are DMA'd
// into LDS by the TDM (double-buffered) and shared by all waves. Uniform
// key-block loop over the block's causal frontier; waves past their own
// frontier skip the compute between barriers. Flash-attention accumulation.

__global__ __launch_bounds__(256) void attn_kernel(
    const _Float16* __restrict__ Qh, const _Float16* __restrict__ Kh,
    const _Float16* __restrict__ Vt, float* __restrict__ Out) {
  __shared__ _Float16 ktile[2][64 * 64];   // 2 x 8 KB  [key][d]
  __shared__ _Float16 vtile[2][64 * 64];   // 2 x 8 KB  [d][key]
  __shared__ _Float16 Plds[8][16][72];     // per-wave P slab
  const int lane  = threadIdx.x & 31;
  const int w     = threadIdx.x >> 5;
  const int lm    = lane & 15;
  const int hf    = lane >> 4;
  const int qblk  = blockIdx.x * 128;
  const int qbase = qblk + w * 16;
  const int h     = blockIdx.y;
  const int b     = blockIdx.z;

  const _Float16* Qb = Qh + (size_t)b * SEQ * D_MODEL + h * D_HEAD;
  const _Float16* Korg = Kh + (size_t)b * SEQ * D_MODEL + h * D_HEAD;
  const _Float16* Vorg = Vt + (size_t)(b * N_HEADS + h) * D_HEAD * SEQ;

  const v16h aQ0 = load_frag(Qb, qbase + lm, D_MODEL, 0,  lane);
  const v16h aQ1 = load_frag(Qb, qbase + lm, D_MODEL, 32, lane);

  v8f o[4] = {vzero(), vzero(), vzero(), vzero()};
  float mrow[8], lrow[8];
#pragma unroll
  for (int r = 0; r < 8; ++r) { mrow[r] = -1e30f; lrow[r] = 0.f; }

  const int kbEndW = (qbase + 15) >> 6;   // this wave's causal frontier
  const int kbEndB = (qblk + 127) >> 6;   // block's frontier (uniform)

#ifdef USE_TDM
  const unsigned kbuf[2] = {(unsigned)(size_t)(&ktile[0][0]),
                            (unsigned)(size_t)(&ktile[1][0])};
  const unsigned vbuf[2] = {(unsigned)(size_t)(&vtile[0][0]),
                            (unsigned)(size_t)(&vtile[1][0])};
  if (w == 0) {
    tdm_load_2d(kbuf[0], Korg, D_MODEL, M_TOT, D_MODEL, 64, 64);
    tdm_load_2d(vbuf[0], Vorg, SEQ, BATCH * N_HEADS * D_HEAD, SEQ, 64, 64);
  }
#else
  coop_copy_tile(ktile[0], Korg, 64, 64, D_MODEL, threadIdx.x, 256);
  coop_copy_tile(vtile[0], Vorg, 64, 64, SEQ, threadIdx.x, 256);
#endif

  for (int kb = 0; kb <= kbEndB; ++kb) {
    __syncthreads();   // everyone done reading the buffer being refilled
#ifdef USE_TDM
    if (w == 0) {
      if (kb + 1 <= kbEndB) {
        tdm_load_2d(kbuf[(kb + 1) & 1], Korg + (size_t)(kb + 1) * 64 * D_MODEL,
                    D_MODEL, M_TOT, D_MODEL, 64, 64);
        tdm_load_2d(vbuf[(kb + 1) & 1], Vorg + (kb + 1) * 64,
                    SEQ, BATCH * N_HEADS * D_HEAD, SEQ, 64, 64);
        __builtin_amdgcn_s_wait_tensorcnt(2);
      } else {
        __builtin_amdgcn_s_wait_tensorcnt(0);
      }
    }
#else
    if (kb + 1 <= kbEndB) {
      coop_copy_tile(ktile[(kb + 1) & 1], Korg + (size_t)(kb + 1) * 64 * D_MODEL,
                     64, 64, D_MODEL, threadIdx.x, 256);
      coop_copy_tile(vtile[(kb + 1) & 1], Vorg + (kb + 1) * 64,
                     64, 64, SEQ, threadIdx.x, 256);
    }
#endif
    __syncthreads();   // tiles ready
    const bool active = (kb <= kbEndW);
    const _Float16* kt = ktile[kb & 1];
    const _Float16* vt = vtile[kb & 1];
    v8f s[4];
    if (active) {
      // ---- S = Q * K^T (16 x 64) from LDS K tile
#pragma unroll
      for (int j = 0; j < 4; ++j) {
        v8f c = vzero();
        c = wmma16(aQ0, load_frag_lds(kt + (j * 16 + lm) * 64 + hf * 8), c);
        c = wmma16(aQ1, load_frag_lds(kt + (j * 16 + lm) * 64 + 32 + hf * 8), c);
        s[j] = c;
      }
      // ---- causal mask (only the diagonal block)
      if (kb == kbEndW) {
#pragma unroll
        for (int j = 0; j < 4; ++j) {
          const int key = kb * 64 + j * 16 + lm;
#pragma unroll
          for (int r = 0; r < 8; ++r)
            if (key > qbase + hf * 8 + r) s[j][r] = -1e30f;
        }
      }
      // ---- online softmax: rows live across 16-lane halves per VGPR
#pragma unroll
      for (int r = 0; r < 8; ++r) {
        float mx = fmaxf(fmaxf(s[0][r], s[1][r]), fmaxf(s[2][r], s[3][r]));
#pragma unroll
        for (int off = 8; off > 0; off >>= 1)
          mx = fmaxf(mx, __shfl_xor(mx, off, 32));
        const float mn   = fmaxf(mrow[r], mx);
        const float corr = __expf(mrow[r] - mn);
        mrow[r] = mn;
        float sum = 0.f;
#pragma unroll
        for (int j = 0; j < 4; ++j) {
          const float p = __expf(s[j][r] - mn);
          s[j][r] = p;
          sum += p;
        }
#pragma unroll
        for (int off = 8; off > 0; off >>= 1)
          sum += __shfl_xor(sum, off, 32);
        lrow[r] = lrow[r] * corr + sum;
#pragma unroll
        for (int t = 0; t < 4; ++t) o[t][r] *= corr;
      }
      // ---- P: C-layout -> A-layout via this wave's LDS slab
#pragma unroll
      for (int j = 0; j < 4; ++j)
#pragma unroll
        for (int r = 0; r < 8; ++r)
          Plds[w][hf * 8 + r][j * 16 + lm] = (_Float16)s[j][r];
    }
    __syncthreads();   // order P write -> read (uniform barrier)
    if (active) {
      const v16h aP0 = load_frag_lds(&Plds[w][lm][hf * 8]);
      const v16h aP1 = load_frag_lds(&Plds[w][lm][32 + hf * 8]);
      // ---- O += P * V from LDS V tile (rows d-major, key-contiguous)
#pragma unroll
      for (int t = 0; t < 4; ++t) {
        const _Float16* vr = vt + (t * 16 + lm) * 64;
        o[t] = wmma16(aP0, load_frag_lds(vr + hf * 8), o[t]);
        o[t] = wmma16(aP1, load_frag_lds(vr + 32 + hf * 8), o[t]);
      }
    }
  }
  // ---- normalize + store [b, t, h*64 + d] f32
#pragma unroll
  for (int r = 0; r < 8; ++r) {
    const float inv = 1.f / lrow[r];
    const int q = qbase + hf * 8 + r;
    float* orow = Out + ((size_t)b * SEQ + q) * D_MODEL + h * D_HEAD + lm;
#pragma unroll
    for (int t = 0; t < 4; ++t)
      orow[t * 16] = o[t][r] * inv;
  }
}

// ---------------------------------------------------------------- launch

extern "C" void kernel_launch(void* const* d_in, const int* in_sizes, int n_in,
                              void* d_out, int out_size, void* d_ws, size_t ws_size,
                              hipStream_t stream) {
  const float* X  = (const float*)d_in[0];
  const float* Wq = (const float*)d_in[1];
  const float* Wk = (const float*)d_in[2];
  const float* Wv = (const float*)d_in[3];
  float* out = (float*)d_out;
  char* ws = (char*)d_ws;

  const size_t SZ_X = (size_t)M_TOT * D_MODEL * sizeof(_Float16);     // 8 MB
  const size_t SZ_W = (size_t)D_MODEL * D_MODEL * sizeof(_Float16);   // 2 MB each
  _Float16* Xh = (_Float16*)(ws);
  _Float16* Wh = (_Float16*)(ws + SZ_X);
  _Float16* Qh = (_Float16*)(ws + SZ_X + 3 * SZ_W);
  _Float16* Kh = (_Float16*)(ws + 2 * SZ_X + 3 * SZ_W);
  _Float16* Vt = (_Float16*)(ws + 3 * SZ_X + 3 * SZ_W);

  const int nX = M_TOT * D_MODEL;
  const int nW = D_MODEL * D_MODEL;
  f32_to_f16<<<nX / 256, 256, 0, stream>>>(X, Xh, nX);
  f32_to_f16<<<nW / 256, 256, 0, stream>>>(Wq, Wh, nW);
  f32_to_f16<<<nW / 256, 256, 0, stream>>>(Wk, Wh + (size_t)nW, nW);
  f32_to_f16<<<nW / 256, 256, 0, stream>>>(Wv, Wh + 2 * (size_t)nW, nW);

  qkv_gemm<<<dim3(M_TOT / 64, D_MODEL / 128, 3), 256, 0, stream>>>(
      Xh, Wh, Qh, Kh, Vt);

  attn_kernel<<<dim3(SEQ / 128, N_HEADS, BATCH), 256, 0, stream>>>(
      Qh, Kh, Vt, out);
}
